// SimilarityGraphBuilder_84138409328872
// MI455X (gfx1250) — matrix operand order repaired and checked by reading, not compiled, not run on hardware
//
#include <hip/hip_runtime.h>
#include <hip/hip_bf16.h>

typedef __attribute__((ext_vector_type(16))) _Float16 v16h;
typedef __attribute__((ext_vector_type(8)))  float    v8f;

#define IN_DIM 512
#define HID    128
#define KTOP   15
#define CWAVES 4          // waves per block in sim kernel (column-split)

__device__ __forceinline__ unsigned ballot32(bool p) {
#if __has_builtin(__builtin_amdgcn_ballot_w32)
    return __builtin_amdgcn_ballot_w32(p);
#else
    return (unsigned)__ballot(p);
#endif
}

// ---------------------------------------------------------------------------
// Zero the [N,N] output (poisoned by harness; mask output is mostly zeros).
// ---------------------------------------------------------------------------
__global__ void zero_kernel(float* __restrict__ out, size_t n) {
    size_t stride = (size_t)gridDim.x * blockDim.x * 4;
    size_t base   = ((size_t)blockIdx.x * blockDim.x + threadIdx.x) * 4;
    for (size_t i = base; i + 3 < n; i += stride) {
        *(float4*)(out + i) = make_float4(0.f, 0.f, 0.f, 0.f);
    }
    if (base == 0) {
        for (size_t i = n & ~(size_t)3; i < n; ++i) out[i] = 0.f;
    }
}

// ---------------------------------------------------------------------------
// Convert W [IN_DIM, HID] f32 -> WhT [HID, IN_DIM] f16 (transposed) so that
// WMMA B-fragments are contiguous 16-half runs per lane.
// ---------------------------------------------------------------------------
__global__ void wprep_kernel(const float* __restrict__ W, _Float16* __restrict__ WhT) {
    int i = blockIdx.x * blockDim.x + threadIdx.x;
    if (i < IN_DIM * HID) {
        int k = i / HID;
        int h = i - k * HID;
        WhT[h * IN_DIM + k] = (_Float16)W[i];
    }
}

// ---------------------------------------------------------------------------
// z = normalize(feat @ W + b), stored as f16 [N, HID].
// One wave per 16-row block: 8 col tiles x 16 k-steps of wmma_f32_16x16x32_f16.
// C layout (ISA 7.12.2): acc[t][v] holds row (v + 8*half), col (t*16 + m).
// ---------------------------------------------------------------------------
__global__ __launch_bounds__(256) void proj_norm_kernel(
    const float* __restrict__ feat, const _Float16* __restrict__ WhT,
    const float* __restrict__ bias, _Float16* __restrict__ zh, int nrows)
{
    const int lane = threadIdx.x & 31;
    const int wave = threadIdx.x >> 5;
    const int rb   = (blockIdx.x * 8 + wave) * 16;
    if (rb >= nrows) return;

    const int m    = lane & 15;
    const int half = lane >> 4;
    const int row  = min(rb + m, nrows - 1);

    v8f acc[8];
#pragma unroll
    for (int t = 0; t < 8; ++t) acc[t] = (v8f){};

    for (int k0 = 0; k0 < IN_DIM; k0 += 32) {
        v16h a;
        const float* fp = feat + (size_t)row * IN_DIM + k0 + half * 16;
#pragma unroll
        for (int h = 0; h < 8; ++h) a[h]     = (_Float16)fp[h];
#pragma unroll
        for (int h = 0; h < 8; ++h) a[8 + h] = (_Float16)fp[8 + h];

#pragma unroll
        for (int t = 0; t < 8; ++t) {
            const v16h bfrag =
                *(const v16h*)(WhT + (size_t)(t * 16 + m) * IN_DIM + k0 + half * 16);
            acc[t] = __builtin_amdgcn_wmma_f32_16x16x32_f16(
                false, a, false, bfrag, (short)0, acc[t], false, false);
        }
    }

#pragma unroll
    for (int t = 0; t < 8; ++t) {
        float bv = bias[t * 16 + m];
#pragma unroll
        for (int v = 0; v < 8; ++v) acc[t][v] += bv;
    }

    float rinv[8];
#pragma unroll
    for (int v = 0; v < 8; ++v) {
        float s = 0.f;
#pragma unroll
        for (int t = 0; t < 8; ++t) s += acc[t][v] * acc[t][v];
        s += __shfl_xor(s, 1, 32);
        s += __shfl_xor(s, 2, 32);
        s += __shfl_xor(s, 4, 32);
        s += __shfl_xor(s, 8, 32);
        rinv[v] = 1.0f / fmaxf(sqrtf(s), 1e-12f);
    }

#pragma unroll
    for (int t = 0; t < 8; ++t) {
#pragma unroll
        for (int v = 0; v < 8; ++v) {
            int r = rb + v + half * 8;
            if (r < nrows)
                zh[(size_t)r * HID + t * 16 + m] = (_Float16)(acc[t][v] * rinv[v]);
        }
    }
}

// ---------------------------------------------------------------------------
// sim = z @ z^T streamed tile-by-tile; fused top-15 per row; scatter output.
// 4 waves per block share one 16-row panel and split the column range; each
// maintains a private top-15 list; lists merged at the end. B-fragment loads
// are double-buffered against the WMMAs. Thresholds cached in registers so
// the common path per tile is 4 WMMAs + 8 compares + 1 ballot.
// ---------------------------------------------------------------------------
__global__ __launch_bounds__(32 * CWAVES) void sim_topk_kernel(
    const _Float16* __restrict__ zh, float* __restrict__ out, int n)
{
    __shared__ float    topv[CWAVES][16][16];
    __shared__ int      topi[CWAVES][16][16];
    __shared__ float    thrL[CWAVES][16];
    __shared__ unsigned cmask[CWAVES][16];
    __shared__ float    cval[CWAVES][16][16];
    __shared__ int      cidx[CWAVES][16][16];

    const int lane = threadIdx.x & 31;
    const int w    = threadIdx.x >> 5;
    const int m    = lane & 15;
    const int half = lane >> 4;
    const int rb   = blockIdx.x * 16;

    if (lane < 16) {
        thrL[w][lane] = -1e30f;
#pragma unroll
        for (int j = 0; j < 16; ++j) {
            topv[w][lane][j] = -1e30f;
            topi[w][lane][j] = -1;
        }
    }
    __threadfence_block();   // wave-local LDS init visibility

    float thrReg[8];
#pragma unroll
    for (int v = 0; v < 8; ++v) thrReg[v] = -1e30f;

    // Preload the 4 A-fragments (entire 16x128 row panel in registers)
    v16h afrag[4];
    {
        const _Float16* ap = zh + (size_t)(rb + m) * HID;
#pragma unroll
        for (int ks = 0; ks < 4; ++ks) {
            int kb = ks * 32 + half * 16;
#pragma unroll
            for (int h = 0; h < 8; ++h) afrag[ks][h]     = ap[kb + h];
#pragma unroll
            for (int h = 0; h < 8; ++h) afrag[ks][8 + h] = ap[kb + 8 + h];
        }
    }

    auto loadB = [&](v16h* buf, int cb) {
        const _Float16* bp = zh + (size_t)(cb + m) * HID;
#pragma unroll
        for (int ks = 0; ks < 4; ++ks)
            buf[ks] = *(const v16h*)(bp + ks * 32 + half * 16);
    };

    auto process = [&](const v16h* buf, int cb) {
        v8f acc = (v8f){};
#pragma unroll
        for (int ks = 0; ks < 4; ++ks)
            acc = __builtin_amdgcn_wmma_f32_16x16x32_f16(
                false, afrag[ks], false, buf[ks], (short)0, acc, false, false);

        // cheap common path: 8 compares, one ballot
        bool anyl = false;
#pragma unroll
        for (int v = 0; v < 8; ++v) anyl |= (acc[v] > thrReg[v]);

        if (ballot32(anyl)) {
            // rare path: record candidates in wave-private LDS, merge serially
            unsigned msks[8];
#pragma unroll
            for (int v = 0; v < 8; ++v) {
                bool pred = acc[v] > thrReg[v];
                msks[v] = ballot32(pred);
                int r = v + half * 8;
                if (pred) { cval[w][r][m] = acc[v]; cidx[w][r][m] = cb + m; }
            }
            if (lane == 0) {
#pragma unroll
                for (int v = 0; v < 8; ++v) {
                    cmask[w][v]     = msks[v] & 0xFFFFu;
                    cmask[w][v + 8] = msks[v] >> 16;
                }
            }
            __threadfence_block();   // wave-local: LDS ops in-order per wave
            if (lane < 16) {
                unsigned mm = cmask[w][lane];
                while (mm) {
                    int b = __ffs(mm) - 1;
                    mm &= mm - 1;
                    float v = cval[w][lane][b];
                    int   i = cidx[w][lane][b];
                    if (v > topv[w][lane][KTOP - 1]) {
                        int p = KTOP - 1;
                        while (p > 0 && topv[w][lane][p - 1] < v) {
                            topv[w][lane][p] = topv[w][lane][p - 1];
                            topi[w][lane][p] = topi[w][lane][p - 1];
                            --p;
                        }
                        topv[w][lane][p] = v;
                        topi[w][lane][p] = i;
                    }
                }
                thrL[w][lane] = topv[w][lane][KTOP - 1];
            }
            __threadfence_block();
#pragma unroll
            for (int v = 0; v < 8; ++v) thrReg[v] = thrL[w][v + half * 8];
        }
    };

    // Column tiles for this wave: w, w+CWAVES, w+2*CWAVES, ... (double-buffered)
    const int ntiles = n >> 4;
    v16h buf0[4], buf1[4];
    int tile = w;
    if (tile < ntiles) {
        loadB(buf0, tile << 4);
        while (tile < ntiles) {
            int t1 = tile + CWAVES;
            if (t1 < ntiles) loadB(buf1, t1 << 4);
            process(buf0, tile << 4);
            int t2 = tile + 2 * CWAVES;
            if (t1 < ntiles) {
                if (t2 < ntiles) loadB(buf0, t2 << 4);
                process(buf1, t1 << 4);
            }
            tile = t2;
        }
    }

    __syncthreads();   // uniform point: all waves done streaming

    // Merge the CWAVES per-wave lists into wave 0's list, then scatter.
    if (threadIdx.x < 16) {
        const int row = threadIdx.x;
        for (int w2 = 1; w2 < CWAVES; ++w2) {
            for (int j = 0; j < KTOP; ++j) {
                float v = topv[w2][row][j];
                if (!(v > topv[0][row][KTOP - 1])) break;   // sorted desc
                int i = topi[w2][row][j];
                int p = KTOP - 1;
                while (p > 0 && topv[0][row][p - 1] < v) {
                    topv[0][row][p] = topv[0][row][p - 1];
                    topi[0][row][p] = topi[0][row][p - 1];
                    --p;
                }
                topv[0][row][p] = v;
                topi[0][row][p] = i;
            }
        }
        const int R = rb + row;
        for (int j = 0; j < KTOP; ++j) {
            int idx = topi[0][row][j];
            float v = topv[0][row][j];
            if (idx >= 0 && idx != R) out[(size_t)R * n + idx] = v;
        }
    }
}

// ---------------------------------------------------------------------------
extern "C" void kernel_launch(void* const* d_in, const int* in_sizes, int n_in,
                              void* d_out, int out_size, void* d_ws, size_t ws_size,
                              hipStream_t stream)
{
    const float* feat = (const float*)d_in[0];
    const float* W    = (const float*)d_in[1];
    const float* bias = (const float*)d_in[2];
    // d_in[3] = k (==15, fixed by reference setup; KTOP is compile-time)

    const int N = in_sizes[0] / IN_DIM;      // 10000

    float*    out = (float*)d_out;
    _Float16* WhT = (_Float16*)d_ws;                                 // 128 KB
    _Float16* zh  = (_Float16*)((char*)d_ws + (size_t)IN_DIM * HID * 2);

    // 1) zero the dense output (only top-k survivors are nonzero)
    zero_kernel<<<2048, 256, 0, stream>>>(out, (size_t)out_size);

    // 2) W f32 -> transposed f16
    wprep_kernel<<<(IN_DIM * HID + 255) / 256, 256, 0, stream>>>(W, WhT);

    // 3) z = normalize(feat @ W + b), f16
    int nwaves  = (N + 15) / 16;             // 625
    int nblocks = (nwaves + 7) / 8;          // 79 blocks x 8 waves
    proj_norm_kernel<<<nblocks, 256, 0, stream>>>(feat, WhT, bias, zh, N);

    // 4) fused sim GEMM + top-15 + scatter (4 column-split waves per row block)
    sim_topk_kernel<<<N / 16, 32 * CWAVES, 0, stream>>>(zh, out, N);
}
